// MetaMambaHistory_70506183131282
// MI455X (gfx1250) — matrix operand order repaired and checked by reading, not compiled, not run on hardware
//
#include <hip/hip_runtime.h>
#include <hip/hip_bf16.h>
#include <math.h>

#define BATCH    2
#define SEQLEN   1024
#define T_TOT    (SEQLEN + 1)            // 1025 (emb row prepended)
#define D_MODEL  1024
#define D_INNER  4096
#define D_STATE  16
#define D_CONV   4
#define DT_RANK  64
#define XCOLS    (DT_RANK + 2 * D_STATE) // 96
#define M_VALID  (BATCH * T_TOT)         // 2050
#define M_PAD    2080                    // 65 * 32 (MT=2 tiles)

typedef __attribute__((ext_vector_type(16))) _Float16 v16h;
typedef __attribute__((ext_vector_type(8)))  _Float16 v8h;
typedef __attribute__((ext_vector_type(8)))  float    v8f;

__device__ __forceinline__ float silu_f(float v) { return v / (1.0f + __expf(-v)); }

// ---------------------------------------------------------------------------
// Build padded xs (f16): row 0 of each batch = emb_table[layer_idx], rest = x.
// Pad rows (m >= M_VALID) are zero so GEMM1 produces exact zeros there.
// ---------------------------------------------------------------------------
__global__ void k_build_xs(const float* __restrict__ x, const int* __restrict__ layer_idx,
                           const float* __restrict__ emb, _Float16* __restrict__ xs_h) {
  const int li = layer_idx[0];
  const size_t total = (size_t)M_PAD * D_MODEL;
  for (size_t i = (size_t)blockIdx.x * blockDim.x + threadIdx.x; i < total;
       i += (size_t)gridDim.x * blockDim.x) {
    int m = (int)(i / D_MODEL);
    int j = (int)(i % D_MODEL);
    float v = 0.0f;
    if (m < M_VALID) {
      int b = m / T_TOT, t = m % T_TOT;
      v = (t == 0) ? emb[(size_t)li * D_MODEL + j]
                   : x[((size_t)b * SEQLEN + (t - 1)) * D_MODEL + j];
    }
    xs_h[i] = (_Float16)v;
  }
}

// ---------------------------------------------------------------------------
// Convert + pack weight W[K,N] (f32 row-major) into WMMA-B fragment-major f16:
//   Wp[((kt*(N/16) + nt)*32 + lane)*16 + h] = W[kt*32 + (h>>3)*16 + (lane>>4)*8 + (h&7)][nt*16 + (lane&15)]
// so a wave's B fragment is one aligned 32-byte load per lane.
// ---------------------------------------------------------------------------
__global__ void k_pack_B(const float* __restrict__ W, _Float16* __restrict__ Wp,
                         int K, int N) {
  const size_t total = (size_t)K * N;
  const int ntiles = N >> 4;
  for (size_t i = (size_t)blockIdx.x * blockDim.x + threadIdx.x; i < total;
       i += (size_t)gridDim.x * blockDim.x) {
    int h    = (int)(i & 15);
    int lane = (int)((i >> 4) & 31);
    size_t tile = i >> 9;
    int kt = (int)(tile / ntiles);
    int nt = (int)(tile % ntiles);
    int k = kt * 32 + ((h >> 3) << 4) + ((lane >> 4) << 3) + (h & 7);
    int n = nt * 16 + (lane & 15);
    Wp[i] = (_Float16)W[(size_t)k * N + n];
  }
}

// ---------------------------------------------------------------------------
// WMMA GEMM: C[M,N] = A[M,K] * Bpack, A row-major f16, Bpack fragment-major.
// Wave tile: (16*MT) x (16*NT). Per 32-deep k-step: 2*MT b128 A-loads,
// 2*NT b128 B-loads, MT*NT v_wmma_f32_16x16x32_f16.
// ---------------------------------------------------------------------------
template <int MT, int NT>
__global__ void k_gemm_wmma(const _Float16* __restrict__ A, const _Float16* __restrict__ Bp,
                            float* __restrict__ C, int M, int N, int K, int Mvalid) {
  const int wid  = blockIdx.x * (blockDim.x >> 5) + (threadIdx.x >> 5);
  const int lane = threadIdx.x & 31;
  const int wavesPerRow = N / (16 * NT);
  const int mt  = wid / wavesPerRow;     // tile row group (16*MT rows)
  const int ntg = wid % wavesPerRow;     // tile col group (16*NT cols)
  if (mt >= (M / (16 * MT))) return;     // uniform per wave

  v8f acc[MT][NT];
#pragma unroll
  for (int a = 0; a < MT; ++a)
#pragma unroll
    for (int b = 0; b < NT; ++b)
#pragma unroll
      for (int e = 0; e < 8; ++e) acc[a][b][e] = 0.0f;

  const int nlo = lane & 15;
  const int kh  = lane >> 4;
  const int ntilesN = N >> 4;
  const int mBase = mt * 16 * MT;

  const _Float16* Arow[MT];
#pragma unroll
  for (int a = 0; a < MT; ++a)
    Arow[a] = A + (size_t)(mBase + a * 16 + nlo) * K + kh * 8;

  for (int k0 = 0; k0 < K; k0 += 32) {
    // A fragments: halves [0..7] = K[k0+kh*8..], [8..15] = K[k0+16+kh*8..]
    v16h af[MT];
#pragma unroll
    for (int a = 0; a < MT; ++a) {
      v8h lo = *(const v8h*)(Arow[a] + k0);
      v8h hi = *(const v8h*)(Arow[a] + k0 + 16);
      af[a] = __builtin_shufflevector(lo, hi, 0, 1, 2, 3, 4, 5, 6, 7,
                                      8, 9, 10, 11, 12, 13, 14, 15);
    }
    const _Float16* Btile =
        Bp + ((size_t)(k0 >> 5) * ntilesN + ntg * NT) * 512 + lane * 16;
#pragma unroll
    for (int b = 0; b < NT; ++b) {
      v16h bf = *(const v16h*)(Btile + (size_t)b * 512);
#pragma unroll
      for (int a = 0; a < MT; ++a) {
        acc[a][b] = __builtin_amdgcn_wmma_f32_16x16x32_f16(
            /*neg_a=*/false, af[a], /*neg_b=*/false, bf,
            /*c_mod=*/(short)0, acc[a][b], /*reuse_a=*/false, /*reuse_b=*/false);
      }
    }
  }

  // C/D f32 layout: lane = N, VGPR v -> M = v + 8*(lane/16)
  const int mh = lane >> 4;
#pragma unroll
  for (int a = 0; a < MT; ++a) {
#pragma unroll
    for (int b = 0; b < NT; ++b) {
      int col = (ntg * NT + b) * 16 + nlo;
#pragma unroll
      for (int v = 0; v < 8; ++v) {
        int row = mBase + a * 16 + mh * 8 + v;
        if (row < Mvalid) C[(size_t)row * N + col] = acc[a][b][v];
      }
    }
  }
}

// ---------------------------------------------------------------------------
// Depthwise causal conv (width 4) + bias + SiLU on xi = xz[:, :, :D_INNER].
// ---------------------------------------------------------------------------
__global__ void k_conv_silu(const float* __restrict__ xz, const float* __restrict__ conv_w,
                            const float* __restrict__ conv_b, float* __restrict__ xh,
                            _Float16* __restrict__ xh_h, _Float16* __restrict__ y_h) {
  const size_t total = (size_t)M_PAD * D_INNER;
  for (size_t i = (size_t)blockIdx.x * blockDim.x + threadIdx.x; i < total;
       i += (size_t)gridDim.x * blockDim.x) {
    int m = (int)(i / D_INNER);
    int d = (int)(i % D_INNER);
    if (m >= M_VALID) {
      xh[i] = 0.0f; xh_h[i] = (_Float16)0.0f; y_h[i] = (_Float16)0.0f;
      continue;
    }
    int b = m / T_TOT, t = m % T_TOT;
    float acc = conv_b[d];
#pragma unroll
    for (int j = 0; j < D_CONV; ++j) {
      int tt = t - (D_CONV - 1) + j;
      if (tt >= 0)
        acc += conv_w[d * D_CONV + j] * xz[(size_t)(b * T_TOT + tt) * (2 * D_INNER) + d];
    }
    float s = silu_f(acc);
    xh[i] = s;
    xh_h[i] = (_Float16)s;
  }
}

// dt_in (cols 0..63 of x_dbl) -> f16 for the dt GEMM
__global__ void k_dt_pre(const float* __restrict__ x_dbl, _Float16* __restrict__ dt_pre_h) {
  const size_t total = (size_t)M_PAD * DT_RANK;
  for (size_t i = (size_t)blockIdx.x * blockDim.x + threadIdx.x; i < total;
       i += (size_t)gridDim.x * blockDim.x) {
    int m = (int)(i / DT_RANK);
    int r = (int)(i % DT_RANK);
    dt_pre_h[i] = (_Float16)x_dbl[(size_t)m * XCOLS + r];
  }
}

// ---------------------------------------------------------------------------
// Selective-scan: one thread per (b, d) channel; 16 h-states in registers.
// ---------------------------------------------------------------------------
__global__ void k_scan(const float* __restrict__ dt_lin, const float* __restrict__ b_dt,
                       const float* __restrict__ A_log, const float* __restrict__ xh,
                       const float* __restrict__ x_dbl, const float* __restrict__ D_skip,
                       const float* __restrict__ xz, _Float16* __restrict__ y_h) {
  const int d = blockIdx.x * blockDim.x + threadIdx.x;   // 0..D_INNER-1
  const int b = blockIdx.y;
  float Aneg[D_STATE], h[D_STATE];
#pragma unroll
  for (int n = 0; n < D_STATE; ++n) {
    Aneg[n] = -__expf(A_log[(size_t)d * D_STATE + n]);
    h[n] = 0.0f;
  }
  const float bd = b_dt[d];
  const float Dd = D_skip[d];
  for (int t = 0; t < T_TOT; ++t) {
    const size_t m = (size_t)b * T_TOT + t;
    float dtl = dt_lin[m * D_INNER + d] + bd;
    float dt  = (dtl > 20.0f) ? dtl : log1pf(__expf(dtl));   // softplus
    float xv  = xh[m * D_INNER + d];
    float dx  = dt * xv;
    const float* Bpt = x_dbl + m * XCOLS + DT_RANK;
    const float* Cpt = Bpt + D_STATE;
    float y = 0.0f;
#pragma unroll
    for (int n = 0; n < D_STATE; ++n) {
      float dA = __expf(dt * Aneg[n]);
      h[n] = h[n] * dA + dx * Bpt[n];
      y += h[n] * Cpt[n];
    }
    y += xv * Dd;                                   // D skip
    float zv = xz[m * (2 * D_INNER) + D_INNER + d];
    y *= silu_f(zv);                                // gate
    y_h[m * D_INNER + d] = (_Float16)y;
  }
}

// ---------------------------------------------------------------------------
extern "C" void kernel_launch(void* const* d_in, const int* in_sizes, int n_in,
                              void* d_out, int out_size, void* d_ws, size_t ws_size,
                              hipStream_t stream) {
  (void)in_sizes; (void)n_in; (void)out_size; (void)ws_size;

  const float* x      = (const float*)d_in[0];
  const int*   lidx   = (const int*)d_in[1];
  const float* emb    = (const float*)d_in[2];
  const float* W_in   = (const float*)d_in[3];
  const float* conv_w = (const float*)d_in[4];
  const float* conv_b = (const float*)d_in[5];
  const float* W_x    = (const float*)d_in[6];
  const float* W_dt   = (const float*)d_in[7];
  const float* b_dt   = (const float*)d_in[8];
  const float* A_log  = (const float*)d_in[9];
  const float* D_skip = (const float*)d_in[10];
  const float* W_out  = (const float*)d_in[11];
  float* out = (float*)d_out;

  char* ws = (char*)d_ws;
  size_t off = 0;
  auto alloc = [&](size_t bytes) -> void* {
    void* p = ws + off;
    off = (off + bytes + 255) & ~(size_t)255;
    return p;
  };

  _Float16* xs_h     = (_Float16*)alloc((size_t)M_PAD * D_MODEL * 2);
  _Float16* W_in_p   = (_Float16*)alloc((size_t)D_MODEL * 2 * D_INNER * 2);
  _Float16* W_x_p    = (_Float16*)alloc((size_t)D_INNER * XCOLS * 2);
  _Float16* W_dt_p   = (_Float16*)alloc((size_t)DT_RANK * D_INNER * 2);
  _Float16* W_out_p  = (_Float16*)alloc((size_t)D_INNER * D_MODEL * 2);
  float*    xz       = (float*)alloc((size_t)M_PAD * 2 * D_INNER * 4);
  float*    xh       = (float*)alloc((size_t)M_PAD * D_INNER * 4);
  _Float16* xh_h     = (_Float16*)alloc((size_t)M_PAD * D_INNER * 2);
  float*    x_dbl    = (float*)alloc((size_t)M_PAD * XCOLS * 4);
  _Float16* dt_pre_h = (_Float16*)alloc((size_t)M_PAD * DT_RANK * 2);
  float*    dt_lin   = (float*)alloc((size_t)M_PAD * D_INNER * 4);
  _Float16* y_h      = (_Float16*)alloc((size_t)M_PAD * D_INNER * 2);

  const int TB = 256;
  auto nb = [](size_t n, int tb) { return (unsigned)((n + tb - 1) / tb); };

  // --- prep: xs (f16) + weight pack (fragment-major f16) ---
  k_build_xs<<<nb((size_t)M_PAD * D_MODEL, TB), TB, 0, stream>>>(x, lidx, emb, xs_h);
  k_pack_B<<<nb((size_t)D_MODEL * 2 * D_INNER, TB), TB, 0, stream>>>(W_in, W_in_p, D_MODEL, 2 * D_INNER);
  k_pack_B<<<nb((size_t)D_INNER * XCOLS, TB), TB, 0, stream>>>(W_x, W_x_p, D_INNER, XCOLS);
  k_pack_B<<<nb((size_t)DT_RANK * D_INNER, TB), TB, 0, stream>>>(W_dt, W_dt_p, DT_RANK, D_INNER);
  k_pack_B<<<nb((size_t)D_INNER * D_MODEL, TB), TB, 0, stream>>>(W_out, W_out_p, D_INNER, D_MODEL);

  // --- GEMM1: xz = xs @ W_in   [2080 x 1024] x [1024 x 8192] ---
  {
    int waves = (M_PAD / 32) * (2 * D_INNER / 64);
    k_gemm_wmma<2, 4><<<(waves + 7) / 8, TB, 0, stream>>>(xs_h, W_in_p, xz, M_PAD, 2 * D_INNER, D_MODEL, M_PAD);
  }

  // --- conv + SiLU -> xh / xh_h ---
  k_conv_silu<<<nb((size_t)M_PAD * D_INNER, TB), TB, 0, stream>>>(xz, conv_w, conv_b, xh, xh_h, y_h);

  // --- GEMM2: x_dbl = xh @ W_x   [2080 x 4096] x [4096 x 96] ---
  {
    int waves = (M_PAD / 32) * (XCOLS / 32);
    k_gemm_wmma<2, 2><<<(waves + 7) / 8, TB, 0, stream>>>(xh_h, W_x_p, x_dbl, M_PAD, XCOLS, D_INNER, M_PAD);
  }

  // --- dt_in -> f16, GEMM3: dt_lin = dt_in @ W_dt  [2080 x 64] x [64 x 4096] ---
  k_dt_pre<<<nb((size_t)M_PAD * DT_RANK, TB), TB, 0, stream>>>(x_dbl, dt_pre_h);
  {
    int waves = (M_PAD / 32) * (D_INNER / 64);
    k_gemm_wmma<2, 4><<<(waves + 7) / 8, TB, 0, stream>>>(dt_pre_h, W_dt_p, dt_lin, M_PAD, D_INNER, DT_RANK, M_PAD);
  }

  // --- selective scan (fp32, 8192 channels in parallel) ---
  {
    dim3 grid(D_INNER / TB, BATCH);
    k_scan<<<grid, TB, 0, stream>>>(dt_lin, b_dt, A_log, xh, x_dbl, D_skip, xz, y_h);
  }

  // --- GEMM4: out = y @ W_out  [2080 x 4096] x [4096 x 1024], store only 2050 rows ---
  {
    int waves = (M_PAD / 32) * (D_MODEL / 64);
    k_gemm_wmma<2, 4><<<(waves + 7) / 8, TB, 0, stream>>>(y_h, W_out_p, out, M_PAD, D_MODEL, D_INNER, M_VALID);
  }
}
